// GreedyDecoder_59700045415113
// MI455X (gfx1250) — matrix operand order repaired and checked by reading, not compiled, not run on hardware
//
#include <hip/hip_runtime.h>
#include <hip/hip_bf16.h>
#include <math.h>
#include <stdint.h>

#define VOCAB 50257
#define EMBD  512
#define HID   1024
#define CTX   1024
#define SEQ   50
#define MAX_STEPS 50
#define BOS 1
#define EOS 2
#define VOCAB_PAD 50304   // multiple of 128 (8 waves * 16 rows per block)
#define XMAX  (EMBD + CTX)   // largest GEMV cols = 1536

typedef __attribute__((ext_vector_type(16))) __bf16 v16bf;
typedef __attribute__((ext_vector_type(8)))  float  v8f;
typedef __attribute__((ext_vector_type(2)))  float  v2f;

__device__ __forceinline__ uint16_t f2bf(float f) {
  __bf16 b = (__bf16)f;           // hardware RNE f32->bf16
  uint16_t u; __builtin_memcpy(&u, &b, 2); return u;
}

// ---------- prologue: Wv fp32 -> bf16 (padded rows zeroed). Makes the full
// weight working set (~139 MB) resident in the 192 MB L2 across all 50 steps.
__global__ void k_cvt_wv(const float* __restrict__ Wv, uint16_t* __restrict__ Wvb) {
  const long long total = (long long)VOCAB_PAD * HID;
  for (long long e = (long long)blockIdx.x * blockDim.x + threadIdx.x; e < total;
       e += (long long)gridDim.x * blockDim.x) {
    int row = (int)(e / HID);
    int col = (int)(e % HID);
    float v = (row < VOCAB) ? Wv[(long long)row * HID + col] : 0.0f;
    Wvb[e] = f2bf(v);
  }
}

__global__ void k_init(const float* __restrict__ E, float* __restrict__ emb,
                       float* __restrict__ scal) {
  int t = threadIdx.x;
  if (t < EMBD) emb[t] = E[(long long)BOS * EMBD + t];
  if (t == 0) { scal[0]=0.f; scal[1]=0.f; scal[2]=0.f; scal[3]=0.f; scal[4]=0.f; }
}

// ---------- small fp32 GEMV (rows not multiple of 16): wave-per-row VALU
__global__ void k_gemv(const float* __restrict__ W, const float* __restrict__ x,
                       const float* __restrict__ b, float* __restrict__ y,
                       int rows, int cols) {
  int lane = threadIdx.x & 31;
  int row  = blockIdx.x * 8 + (threadIdx.x >> 5);
  if (row >= rows) return;
  const float* wr = W + (long long)row * cols;
  float acc = 0.f;
  for (int c = lane; c < cols; c += 32) acc += wr[c] * x[c];
  #pragma unroll
  for (int off = 16; off >= 1; off >>= 1) acc += __shfl_xor(acc, off, 32);
  if (lane == 0) y[row] = acc + (b ? b[row] : 0.0f);
}

// ---------- fp32 GEMV on the matrix pipe: V_WMMA_F32_16X16X4_F32.
// A = 16x4 fp32 tile of W (ISA layout: lanes 0-15 hold K=0,1; lanes 16-31
// hold K=2,3 -> one 8-byte load per lane per K-step). B = x broadcast into
// all 16 columns from LDS. Identical fp32 accumulate numerics as VALU FMA.
__global__ void __launch_bounds__(256)
k_gemv_wmma_f32(const float* __restrict__ W, const float* __restrict__ x,
                const float* __restrict__ b, float* __restrict__ y,
                int rows, int cols) {     // rows % 16 == 0, cols % 4 == 0
  __shared__ float xs[XMAX];
  int tid = threadIdx.x;
  for (int i = tid; i < cols; i += 256) xs[i] = x[i];
  __syncthreads();
  int wave = tid >> 5;
  int lane = tid & 31;
  int half = lane >> 4;                        // 0 -> K 0,1 ; 1 -> K 2,3
  int row0 = (blockIdx.x * 8 + wave) * 16;
  if (row0 >= rows) return;                    // wave-uniform exit, EXEC stays full
  const float* wr = W + (size_t)(row0 + (lane & 15)) * cols + half * 2;
  v8f c = {};
  for (int kc = 0; kc < cols; kc += 4) {
    v2f a = *(const v2f*)(wr + kc);
    v2f bb = *(const v2f*)(xs + kc + half * 2);
    c = __builtin_amdgcn_wmma_f32_16x16x4_f32(false, a, false, bb,
                                              (short)0, c, false, false);
  }
  // C layout: lane 0 holds M=0..7, lane 16 holds M=8..15 (columns identical)
  if ((lane & 15) == 0) {
    int mbase = row0 + (half ? 8 : 0);
    #pragma unroll
    for (int m = 0; m < 8; ++m) {
      int r = mbase + m;
      y[r] = c[m] + (b ? b[r] : 0.0f);
    }
  }
}

__global__ void k_softmax_scores(float* __restrict__ s) {
  float m = -INFINITY;
  for (int i = 0; i < SEQ; ++i) m = fmaxf(m, s[i]);
  float e[SEQ]; float sum = 0.f;
  for (int i = 0; i < SEQ; ++i) { e[i] = expf(s[i] - m); sum += e[i]; }
  float inv = 1.0f / sum;
  for (int i = 0; i < SEQ; ++i) s[i] = e[i] * inv;
}

// x = concat(emb, attn @ enc)
__global__ void k_build_x(const float* __restrict__ emb, const float* __restrict__ attn,
                          const float* __restrict__ enc, float* __restrict__ x) {
  int c = blockIdx.x * blockDim.x + threadIdx.x;
  if (c < EMBD) {
    x[c] = emb[c];
  } else if (c < EMBD + CTX) {
    int cc = c - EMBD;
    float acc = 0.f;
    for (int s = 0; s < SEQ; ++s) acc += attn[s] * enc[(long long)s * CTX + cc];
    x[c] = acc;
  }
}

__global__ void k_gru(const float* __restrict__ gi, const float* __restrict__ gh,
                      const float* __restrict__ h, float* __restrict__ h_new) {
  int j = blockIdx.x * blockDim.x + threadIdx.x;
  if (j >= HID) return;
  float r = 1.f / (1.f + expf(-(gi[j]         + gh[j])));
  float z = 1.f / (1.f + expf(-(gi[HID + j]   + gh[HID + j])));
  float n = tanhf(gi[2*HID + j] + r * gh[2*HID + j]);
  h_new[j] = (1.f - z) * n + z * h[j];
}

// ---------- vocab projection via WMMA bf16: A = 16x32 tile of Wv,
// B = h broadcast into all 16 columns (every column identical).
__global__ void __launch_bounds__(256)
k_logits_wmma(const uint16_t* __restrict__ Wvb, const float* __restrict__ hvec,
              const float* __restrict__ bv, float* __restrict__ logits) {
  __shared__ uint32_t hp[HID / 2];             // h as packed bf16 pairs
  int tid = threadIdx.x;
  for (int i = tid; i < HID / 2; i += 256) {   // uniform trip count: no divergence
    uint32_t lo = f2bf(hvec[2*i]);
    uint32_t hi = f2bf(hvec[2*i + 1]);
    hp[i] = lo | (hi << 16);
  }
  __syncthreads();

  int wave = tid >> 5;
  int lane = tid & 31;
  int half = lane >> 4;                        // 0: K 0..7 / 16..23, 1: K 8..15 / 24..31
  int row0 = (blockIdx.x * 8 + wave) * 16;
  int arow = row0 + (lane & 15);
  const uint32_t* wr = (const uint32_t*)(Wvb + (long long)arow * HID);

  union F { v16bf v; uint32_t u[8]; uint4 q[2]; };
  v8f c = {};
  for (int kc = 0; kc < HID; kc += 32) {
    F a, b;
    int d0 = (kc >> 1) + half * 4;             // dword index of K = kc + half*8
    a.q[0] = *(const uint4*)(wr + d0);         // K = kc+off .. +7
    a.q[1] = *(const uint4*)(wr + d0 + 8);     // K = kc+16+off .. +7
    int bi = (kc >> 1) + half * 4;             // B: K pairs, same for all 16 cols
    #pragma unroll
    for (int j = 0; j < 8; ++j) b.u[j] = hp[bi + j + (j >= 4 ? 4 : 0)];
    c = __builtin_amdgcn_wmma_f32_16x16x32_bf16(false, a.v, false, b.v,
                                                (short)0, c, false, false);
  }
  // C layout: lane 0 holds M=0..7 (any column), lane 16 holds M=8..15
  if ((lane & 15) == 0) {
    int mbase = row0 + (half ? 8 : 0);
    #pragma unroll
    for (int m = 0; m < 8; ++m) {
      int r = mbase + m;
      if (r < VOCAB) logits[r] = c[m] + bv[r];
    }
  }
}

// ---------- argmax + logsumexp over 50257 (single 1024-thread block)
__global__ void __launch_bounds__(1024)
k_argmax_lse(const float* __restrict__ logits, float* __restrict__ scal,
             float* __restrict__ out_idx, float* __restrict__ out_valid, int step) {
  __shared__ float sm[1024];
  __shared__ int   si[1024];
  int t = threadIdx.x;
  float bm = -INFINITY; int bi = 0x7fffffff;
  for (int i = t; i < VOCAB; i += 1024) {
    float v = logits[i];
    if (v > bm || (v == bm && i < bi)) { bm = v; bi = i; }
  }
  sm[t] = bm; si[t] = bi;
  __syncthreads();
  for (int s = 512; s > 0; s >>= 1) {
    if (t < s) {
      float v = sm[t + s]; int i2 = si[t + s];
      if (v > sm[t] || (v == sm[t] && i2 < si[t])) { sm[t] = v; si[t] = i2; }
    }
    __syncthreads();
  }
  float maxv = sm[0];
  int   idx  = si[0];
  __syncthreads();
  float p = 0.f;
  for (int i = t; i < VOCAB; i += 1024) p += expf(logits[i] - maxv);
  sm[t] = p;
  __syncthreads();
  for (int s = 512; s > 0; s >>= 1) {
    if (t < s) sm[t] += sm[t + s];
    __syncthreads();
  }
  if (t == 0) {
    float lse   = maxv + logf(sm[0]);
    int valid   = (scal[0] == 0.0f);
    int is_eos  = (idx == EOS);
    int emit    = valid && !is_eos;
    out_idx[step]   = emit  ? (float)idx : -1.0f;
    out_valid[step] = valid ? 1.0f : 0.0f;
    scal[1] = lse;
    scal[2] = (float)idx;
    scal[3] = valid ? 1.0f : 0.0f;
    scal[4] = emit  ? 1.0f : 0.0f;
    scal[0] = (!valid || is_eos) ? 1.0f : 0.0f;   // done |= is_eos
  }
}

__global__ void k_write_logp(const float* __restrict__ logits, const float* __restrict__ scal,
                             float* __restrict__ out_logp) {
  int i = blockIdx.x * blockDim.x + threadIdx.x;
  if (i >= VOCAB) return;
  out_logp[i] = (scal[3] != 0.0f) ? (logits[i] - scal[1]) : 0.0f;
}

__global__ void k_update(const float* __restrict__ E, const float* __restrict__ scal,
                         const float* __restrict__ h_new, float* __restrict__ h,
                         float* __restrict__ emb) {
  int j = threadIdx.x;  // 1024 threads
  if (scal[3] != 0.0f && j < HID) h[j] = h_new[j];
  if (scal[4] != 0.0f && j < EMBD) {
    int idx = (int)scal[2];
    emb[j] = E[(long long)idx * EMBD + j];
  }
}

extern "C" void kernel_launch(void* const* d_in, const int* in_sizes, int n_in,
                              void* d_out, int out_size, void* d_ws, size_t ws_size,
                              hipStream_t stream) {
  const float* enc_states = (const float*)d_in[0];  // (SEQ,1,CTX) contiguous
  const float* enc_hidden = (const float*)d_in[1];
  const float* E      = (const float*)d_in[2];
  const float* W_init = (const float*)d_in[3];
  const float* b_init = (const float*)d_in[4];
  const float* Wa     = (const float*)d_in[5];
  const float* ba     = (const float*)d_in[6];
  const float* W_ih   = (const float*)d_in[7];
  const float* W_hh   = (const float*)d_in[8];
  const float* b_ih   = (const float*)d_in[9];
  const float* b_hh   = (const float*)d_in[10];
  const float* Wv     = (const float*)d_in[11];
  const float* bv     = (const float*)d_in[12];

  const size_t WVB_BYTES = (size_t)VOCAB_PAD * HID * 2;
  const size_t FLT_CNT = (size_t)HID*2 + EMBD + CTX + 64 + (EMBD+CTX) + 6*HID + VOCAB_PAD + 16;
  if (ws_size < WVB_BYTES + FLT_CNT * sizeof(float)) return;

  uint16_t* Wvb = (uint16_t*)d_ws;
  float* f = (float*)((char*)d_ws + WVB_BYTES);
  float* h      = f;  f += HID;
  float* h_new  = f;  f += HID;
  float* emb    = f;  f += EMBD;
  float* ph     = f;  f += CTX;
  float* scores = f;  f += 64;
  float* x      = f;  f += (EMBD + CTX);
  float* gi     = f;  f += 3 * HID;
  float* gh     = f;  f += 3 * HID;
  float* logits = f;  f += VOCAB_PAD;
  float* scal   = f;  f += 16;

  float* out       = (float*)d_out;                 // (indices, logps, valid) flat
  float* out_idx   = out;
  float* out_logp  = out + MAX_STEPS;
  float* out_valid = out + MAX_STEPS + (size_t)MAX_STEPS * VOCAB;

  k_cvt_wv<<<2048, 256, 0, stream>>>(Wv, Wvb);
  k_init<<<1, 512, 0, stream>>>(E, emb, scal);
  k_gemv_wmma_f32<<<HID/128, 256, 0, stream>>>(W_init, enc_hidden, b_init, h, HID, CTX);

  for (int step = 0; step < MAX_STEPS; ++step) {
    k_gemv_wmma_f32<<<CTX/128, 256, 0, stream>>>(Wa, h, ba, ph, CTX, HID);
    k_gemv<<<(SEQ+7)/8, 256, 0, stream>>>(enc_states, ph, nullptr, scores, SEQ, CTX);
    k_softmax_scores<<<1, 1, 0, stream>>>(scores);
    k_build_x<<<(EMBD+CTX)/256, 256, 0, stream>>>(emb, scores, enc_states, x);
    k_gemv_wmma_f32<<<(3*HID)/128, 256, 0, stream>>>(W_ih, x, b_ih, gi, 3*HID, EMBD+CTX);
    k_gemv_wmma_f32<<<(3*HID)/128, 256, 0, stream>>>(W_hh, h, b_hh, gh, 3*HID, HID);
    k_gru<<<HID/256, 256, 0, stream>>>(gi, gh, h, h_new);
    k_logits_wmma<<<VOCAB_PAD/128, 256, 0, stream>>>(Wvb, h_new, bv, logits);
    k_argmax_lse<<<1, 1024, 0, stream>>>(logits, scal, out_idx, out_valid, step);
    k_write_logp<<<(VOCAB+255)/256, 256, 0, stream>>>(logits, scal,
                                                      out_logp + (size_t)step * VOCAB);
    k_update<<<1, 1024, 0, stream>>>(E, scal, h_new, h, emb);
  }
}
